// GraphConvolutionLayer_19722489823522
// MI455X (gfx1250) — matrix operand order repaired and checked by reading, not compiled, not run on hardware
//
#include <hip/hip_runtime.h>

typedef __bf16 v8bf  __attribute__((ext_vector_type(8)));
typedef __bf16 v16bf __attribute__((ext_vector_type(16)));
typedef float  v8f   __attribute__((ext_vector_type(8)));
typedef float  v4f   __attribute__((ext_vector_type(4)));
typedef unsigned short us4 __attribute__((ext_vector_type(4)));

__device__ __forceinline__ unsigned short f32_to_bf16_rne(float f) {
    union { float f; unsigned u; } cv; cv.f = f;
    unsigned u = cv.u + 0x7FFFu + ((cv.u >> 16) & 1u);   // round-to-nearest-even
    return (unsigned short)(u >> 16);
}

// Tile geometry
constexpr int BM = 128;          // rows per workgroup
constexpr int BN = 128;          // cols per workgroup
constexpr int KB = 64;           // K per main-loop iteration (2 WMMA K-steps)
constexpr int KPAD = KB + 8;     // LDS row pitch (bf16 elems) to stagger banks

// A-matrix 16x32 bf16 fragment (ISA §7.12.2 16-bit A table):
// lanes 0-15: K 0-7 in v0..3, K 16-23 in v4..7; lanes 16-31: K 8-15 / 24-31.
__device__ __forceinline__ v16bf load_frag_A(const unsigned short* rowbase,
                                             int ks, int khalf) {
    const v8bf lo = *(const v8bf*)(rowbase + ks + khalf * 8);
    const v8bf hi = *(const v8bf*)(rowbase + ks + 16 + khalf * 8);
    return __builtin_shufflevector(lo, hi,
        0, 1, 2, 3, 4, 5, 6, 7, 8, 9, 10, 11, 12, 13, 14, 15);
}

// B-matrix 32x16 bf16 fragment (same K order as the A fragment):
// lanes 0-15: K 0-7 in v0..3, K 16-23 in v4..7; lanes 16-31: K 8-15 / 24-31.
// Two 16-byte chunks per lane from the transposed LDS tile.
__device__ __forceinline__ v16bf load_frag_B(const unsigned short* rowbase,
                                             int ks, int khalf) {
    const v8bf lo = *(const v8bf*)(rowbase + ks + khalf * 8);
    const v8bf hi = *(const v8bf*)(rowbase + ks + 16 + khalf * 8);
    return __builtin_shufflevector(lo, hi,
        0, 1, 2, 3, 4, 5, 6, 7, 8, 9, 10, 11, 12, 13, 14, 15);
}

// C = relu?( (sum over PLANES of A_p) @ B )   with bf16 WMMA, f32 accumulate.
// A: f32 [PLANES x M x lda] (planeStride elems between planes)
// B: [K x ldb], f32 (converted) or bf16 per B_IS_BF16
// C: f32 (+ReLU) if RELU_F32_OUT else bf16
template<int PLANES, bool B_IS_BF16, bool RELU_F32_OUT>
__global__ __launch_bounds__(256)
void gcn_gemm(const float* __restrict__ A, size_t planeStride, int lda,
              const void* __restrict__ Bsrc, int ldb,
              void* __restrict__ Cdst, int ldc, int Kdim) {
    __shared__ unsigned short Alds[2][BM][KPAD];
    __shared__ unsigned short Blds[2][BN][KPAD];

    const int tid = threadIdx.x;
    const int m0 = blockIdx.y * BM;
    const int n0 = blockIdx.x * BN;

    const int wave  = tid >> 5;          // 8 waves: 2 (M) x 4 (N)
    const int lane  = tid & 31;
    const int wm    = wave >> 2;         // 0..1 -> 64-row band
    const int wn    = wave & 3;          // 0..3 -> 32-col band
    const int lrow  = lane & 15;
    const int khalf = lane >> 4;

    v8f acc[4][2];
#pragma unroll
    for (int tm = 0; tm < 4; ++tm)
#pragma unroll
        for (int tn = 0; tn < 2; ++tn)
            acc[tm][tn] = v8f{0.f, 0.f, 0.f, 0.f, 0.f, 0.f, 0.f, 0.f};

    const int T = Kdim / KB;

    // --- tile staging: global f32 -> (plane sum) -> bf16 -> LDS ---
    auto stage = [&](int t, int buf) {
        const int k0 = t * KB;
        // A tile: 128 rows x 64 cols; 16 threads x float4 cover one row.
        const int c4    = tid & 15;      // float4 index within row
        const int rbase = tid >> 4;      // 0..15
#pragma unroll
        for (int pass = 0; pass < 8; ++pass) {
            const int row = rbase + pass * 16;
            v4f s = {0.f, 0.f, 0.f, 0.f};
#pragma unroll
            for (int p = 0; p < PLANES; ++p) {
                const float* ap = A + (size_t)p * planeStride
                                    + (size_t)(m0 + row) * lda + k0 + c4 * 4;
                s += *(const v4f*)ap;
                // L2 prefetch of the tile two iterations ahead (the double
                // buffer already covers distance 1 with real loads).
                if (pass == 0 && t + 1 < Kdim / KB)
                    __builtin_prefetch(ap + KB, 0, 0);
            }
            us4 pk;
            pk.x = f32_to_bf16_rne(s.x);
            pk.y = f32_to_bf16_rne(s.y);
            pk.z = f32_to_bf16_rne(s.z);
            pk.w = f32_to_bf16_rne(s.w);
            *(us4*)&Alds[buf][row][c4 * 4] = pk;
        }
        // B tile: stage transposed -> Blds[n][k]. Lane owns column n, packs
        // 4 consecutive K values into one 8-byte LDS store.
        const int nn = tid & 127;
        const int kc = tid >> 7;         // 0..1 -> K half
#pragma unroll
        for (int j = 0; j < 8; ++j) {
            const int k = kc * 32 + j * 4;
            us4 pk;
#pragma unroll
            for (int i = 0; i < 4; ++i) {
                unsigned short e;
                if constexpr (B_IS_BF16) {
                    e = ((const unsigned short*)Bsrc)[(size_t)(k0 + k + i) * ldb + n0 + nn];
                } else {
                    e = f32_to_bf16_rne(((const float*)Bsrc)[(size_t)(k0 + k + i) * ldb + n0 + nn]);
                }
                pk[i] = e;
            }
            *(us4*)&Blds[buf][nn][k] = pk;
        }
    };

    stage(0, 0);
    __syncthreads();

    for (int t = 0; t < T; ++t) {
        const int buf = t & 1;
        if (t + 1 < T) stage(t + 1, buf ^ 1);   // fill other buffer

#pragma unroll
        for (int ks = 0; ks < KB; ks += 32) {
            v16bf afrag[4], bfrag[2];
#pragma unroll
            for (int tn = 0; tn < 2; ++tn)
                bfrag[tn] = load_frag_B(&Blds[buf][wn * 32 + tn * 16 + lrow][0], ks, khalf);
#pragma unroll
            for (int tm = 0; tm < 4; ++tm)
                afrag[tm] = load_frag_A(&Alds[buf][wm * 64 + tm * 16 + lrow][0], ks, khalf);
#pragma unroll
            for (int tm = 0; tm < 4; ++tm)
#pragma unroll
                for (int tn = 0; tn < 2; ++tn)
                    acc[tm][tn] = __builtin_amdgcn_wmma_f32_16x16x32_bf16(
                        false, afrag[tm], false, bfrag[tn],
                        (short)0, acc[tm][tn], false, false);
        }
        __syncthreads();
    }

    // Epilogue. C/D layout: VGPR v, lanes 0-15: M=v, lanes 16-31: M=v+8; N=lane&15.
#pragma unroll
    for (int tm = 0; tm < 4; ++tm)
#pragma unroll
        for (int tn = 0; tn < 2; ++tn)
#pragma unroll
            for (int v = 0; v < 8; ++v) {
                const int grow = m0 + wm * 64 + tm * 16 + khalf * 8 + v;
                const int gcol = n0 + wn * 32 + tn * 16 + lrow;
                const float val = acc[tm][tn][v];
                if constexpr (RELU_F32_OUT) {
                    ((float*)Cdst)[(size_t)grow * ldc + gcol] = fmaxf(val, 0.f);
                } else {
                    ((unsigned short*)Cdst)[(size_t)grow * ldc + gcol] = f32_to_bf16_rne(val);
                }
            }
}

extern "C" void kernel_launch(void* const* d_in, const int* in_sizes, int n_in,
                              void* d_out, int out_size, void* d_ws, size_t ws_size,
                              hipStream_t stream) {
    constexpr int N = 8192, F_IN = 512, F_OUT = 512;
    const float* x   = (const float*)d_in[0];   // [N, F_IN]
    const float* adj = (const float*)d_in[1];   // [4, N, N]
    const float* W   = (const float*)d_in[2];   // [F_IN, F_OUT]
    unsigned short* h = (unsigned short*)d_ws;  // bf16 [N, F_OUT]
    float* out = (float*)d_out;                 // f32 [N, F_OUT]

    dim3 blk(256);
    // Phase 1: h = bf16(x @ W)
    gcn_gemm<1, false, false><<<dim3(F_OUT / BN, N / BM), blk, 0, stream>>>(
        x, 0, F_IN, (const void*)W, F_OUT, (void*)h, F_OUT, F_IN);
    // Phase 2: out = relu( (sum_k adj_k) @ h ), plane-sum fused into staging
    gcn_gemm<4, true, true><<<dim3(F_OUT / BN, N / BM), blk, 0, stream>>>(
        adj, (size_t)N * N, N, (const void*)h, F_OUT, (void*)out, F_OUT, N);
}
